// SPINN4d_8632884265168
// MI455X (gfx1250) — matrix operand order repaired and checked by reading, not compiled, not run on hardware
//
#include <hip/hip_runtime.h>
#include <hip/hip_bf16.h>
#include <math.h>

#define N96 96
#define H64 64
#define R32 32
#define NP  9216            // 96*96 pairs per side
#define TILES 576           // 9216 / 16

typedef __attribute__((ext_vector_type(16))) _Float16 v16h;
typedef __attribute__((ext_vector_type(8)))  float    v8f;
typedef __attribute__((ext_vector_type(4)))  float    v4f;

// ---------------------------------------------------------------------------
// Kernel 1: per-axis MLP. 4 blocks (one per axis), 96 threads (one per point).
// F[i][n][r] = (coords[i][n] -> tanh -> tanh -> linear)[r],  r in [0,32)
// ---------------------------------------------------------------------------
__global__ __launch_bounds__(96)
void spinn_mlp_kernel(const float* __restrict__ t, const float* __restrict__ x,
                      const float* __restrict__ y, const float* __restrict__ z,
                      const float* __restrict__ W1, const float* __restrict__ b1,
                      const float* __restrict__ W2, const float* __restrict__ b2,
                      const float* __restrict__ W3, const float* __restrict__ b3,
                      float* __restrict__ F /* [4][96][32] */) {
    __shared__ float sW2[H64 * H64];   // 16 KB
    __shared__ float sW3[H64 * R32];   // 8 KB
    __shared__ float sW1[H64], sb1[H64], sb2[H64], sb3[R32];

    const int i   = blockIdx.x;        // axis
    const int tid = threadIdx.x;       // point 0..95

    for (int idx = tid; idx < H64 * H64; idx += 96) sW2[idx] = W2[i * H64 * H64 + idx];
    for (int idx = tid; idx < H64 * R32; idx += 96) sW3[idx] = W3[i * H64 * R32 + idx];
    if (tid < H64) {
        sW1[tid] = W1[i * H64 + tid];
        sb1[tid] = b1[i * H64 + tid];
        sb2[tid] = b2[i * H64 + tid];
    }
    if (tid < R32) sb3[tid] = b3[i * R32 + tid];
    __syncthreads();

    const float* coords[4] = {t, x, y, z};
    const float c = coords[i][tid];

    float h1[H64];
#pragma unroll
    for (int h = 0; h < H64; ++h) h1[h] = tanhf(c * sW1[h] + sb1[h]);

    float o[R32];
#pragma unroll
    for (int r = 0; r < R32; ++r) o[r] = sb3[r];

    for (int j = 0; j < H64; ++j) {            // second layer column j
        float acc = sb2[j];
#pragma unroll
        for (int h = 0; h < H64; ++h) acc += h1[h] * sW2[h * H64 + j];
        const float a = tanhf(acc);
#pragma unroll
        for (int r = 0; r < R32; ++r) o[r] += a * sW3[j * R32 + r];
    }

#pragma unroll
    for (int r = 0; r < R32; ++r) F[(i * N96 + tid) * R32 + r] = o[r];
}

// ---------------------------------------------------------------------------
// Kernel 2: pairwise products, fp32 -> fp16.
//   TX [t*96+x][f] = F0[t][f] * F1[x][f]     (A operand, row-major, 64 B rows)
//   YZ [y*96+z][f] = F2[y][f] * F3[z][f]     (B operand stored K-transposed)
// ---------------------------------------------------------------------------
__global__ __launch_bounds__(256)
void spinn_outer_kernel(const float* __restrict__ F,
                        _Float16* __restrict__ TX, _Float16* __restrict__ YZ) {
    const int gid  = blockIdx.x * blockDim.x + threadIdx.x; // 2*9216*32 total
    const int f    = gid & 31;
    int       pair = gid >> 5;                               // 0..18431
    const int m    = pair >= NP ? 1 : 0;                     // 0: TX, 1: YZ
    pair -= m * NP;
    const int u = pair / N96, v = pair - u * N96;
    const float a = F[((2 * m + 0) * N96 + u) * R32 + f];
    const float b = F[((2 * m + 1) * N96 + v) * R32 + f];
    (m ? YZ : TX)[pair * R32 + f] = (_Float16)(a * b);
}

// ---------------------------------------------------------------------------
// Kernel 3: pred = TX(9216x32) @ YZ^T(32x9216) via v_wmma_f32_16x16x32_f16.
// One wave = one 16-row tile x 8 consecutive 16-col tiles (A frag reused).
// 41,472 waves total; stores are the bandwidth-bound 340 MB stream.
// ---------------------------------------------------------------------------
__global__ __launch_bounds__(256)
void spinn_cp_gemm_kernel(const _Float16* __restrict__ TX,
                          const _Float16* __restrict__ YZ,
                          float* __restrict__ out) {
    const int lane = threadIdx.x & 31;
    const int warp = threadIdx.x >> 5;
    const int wg   = blockIdx.x * 8 + warp;        // 0..41471
    const int rtile  = wg / 72;                    // 0..575
    const int cgroup = wg - rtile * 72;            // 0..71
    const int rowbase  = rtile * 16;
    const int colbase0 = cgroup * 128;
    const int half = lane >> 4;
    const int l15  = lane & 15;

    union Frag { v16h h; v4f f[2]; };

    // A fragment: row = rowbase + (lane&15); elements [K0..7,K16..23] (lanes 0-15)
    // or [K8..15,K24..31] (lanes 16-31)  -> two b128 loads.
    const v4f* Arow = (const v4f*)(TX + (size_t)(rowbase + l15) * R32);
    Frag a;
    a.f[0] = Arow[half];         // byte offset half*16
    a.f[1] = Arow[2 + half];     // byte offset 32 + half*16

#pragma unroll
    for (int ct = 0; ct < 8; ++ct) {
        const int colbase = colbase0 + ct * 16;

        // B fragment: column n = colbase + (lane&15), K = [K0..7,K16..23] (lanes 0-15)
        // or [K8..15,K24..31] (lanes 16-31) in the K-transposed YZ row -> two b128 loads.
        const v4f* Brow = (const v4f*)(YZ + (size_t)(colbase + l15) * R32);
        Frag b;
        b.f[0] = Brow[half];
        b.f[1] = Brow[2 + half];

        v8f acc = {0.f, 0.f, 0.f, 0.f, 0.f, 0.f, 0.f, 0.f};
        acc = __builtin_amdgcn_wmma_f32_16x16x32_f16(
            /*neg_a=*/false, a.h, /*neg_b=*/false, b.h,
            /*c_mod=*/(short)0, acc, /*reuse_a=*/false, /*reuse_b=*/false);

        // D layout: VGPR i -> row rowbase + half*8 + i, col colbase + (lane&15)
        float* o = out + (size_t)(rowbase + half * 8) * NP + colbase + l15;
#pragma unroll
        for (int i = 0; i < 8; ++i)
            __builtin_nontemporal_store(acc[i], o + (size_t)i * NP);
    }
}

// ---------------------------------------------------------------------------
extern "C" void kernel_launch(void* const* d_in, const int* in_sizes, int n_in,
                              void* d_out, int out_size, void* d_ws, size_t ws_size,
                              hipStream_t stream) {
    const float* t  = (const float*)d_in[0];
    const float* x  = (const float*)d_in[1];
    const float* y  = (const float*)d_in[2];
    const float* z  = (const float*)d_in[3];
    const float* W1 = (const float*)d_in[4];
    const float* b1 = (const float*)d_in[5];
    const float* W2 = (const float*)d_in[6];
    const float* b2 = (const float*)d_in[7];
    const float* W3 = (const float*)d_in[8];
    const float* b3 = (const float*)d_in[9];
    float* out = (float*)d_out;

    char* ws = (char*)d_ws;
    float*    F  = (float*)ws;                                  // 4*96*32*4  = 49152 B
    _Float16* TX = (_Float16*)(ws + 49152);                     // 9216*32*2  = 589824 B
    _Float16* YZ = (_Float16*)(ws + 49152 + 589824);            // 9216*32*2

    spinn_mlp_kernel<<<4, 96, 0, stream>>>(t, x, y, z, W1, b1, W2, b2, W3, b3, F);

    // 2 * 9216 * 32 = 589824 elements -> 2304 blocks of 256
    spinn_outer_kernel<<<2304, 256, 0, stream>>>(F, TX, YZ);

    // 576 row tiles * 72 col groups = 41472 waves -> 5184 blocks of 8 waves
    spinn_cp_gemm_kernel<<<5184, 256, 0, stream>>>(TX, YZ, out);
}
